// EvoformerNoTriMul_83760452206850
// MI455X (gfx1250) — matrix operand order repaired and checked by reading, not compiled, not run on hardware
//
#include <hip/hip_runtime.h>
#include <stdint.h>

// ---------------- model dims ----------------
#define Bc 2
#define Lc 384
#define AAc 48
#define Hc 256
#define Pc 64
#define NHc 4
#define NBc 64
#define HDc 64
#define NLAY 4
#define NRECc 3
#define MAXRELc 32
#define BLc (Bc * Lc)        // 768
#define BLLc 294912          // B*L*L

typedef __attribute__((ext_vector_type(16))) __bf16 v16bf;
typedef __attribute__((ext_vector_type(8)))  __bf16 v8bf;
typedef __attribute__((ext_vector_type(8)))  float  v8f;
typedef __attribute__((ext_vector_type(4)))  unsigned int u32x4;
typedef __attribute__((ext_vector_type(4)))  int    i32x4;
typedef __attribute__((ext_vector_type(8)))  int    i32x8;

__device__ __forceinline__ uint16_t f2b(float f) {
  uint32_t u = __float_as_uint(f);
  uint32_t r = u + 0x7FFFu + ((u >> 16) & 1u);
  return (uint16_t)(r >> 16);
}
__device__ __forceinline__ float b2f(uint16_t b) {
  return __uint_as_float(((uint32_t)b) << 16);
}
__device__ __forceinline__ float gelu_ex(float x) {
  return 0.5f * x * (1.0f + erff(x * 0.70710678118654752f));
}

// ---------------- WMMA GEMM ----------------
// C[M,N] = ep( A[M,K]_bf16 @ B[N,K]_bf16^T )   (B stored as W^T row-major [N,K])
// epilogue: v = acc*scale + bias[n] + addm[m,n]; if(act) gelu; v += resid[m,n];
// z-batching: offsets = (z/zdiv)*s?b + (z%zdiv)*s?h
struct GemmP {
  const uint16_t* A; int lda; long long sAb, sAh;
  const uint16_t* Bm; int ldb; long long sBb, sBh;
  float* outF; uint16_t* outB; int ldc; long long sCb, sCh;
  const float* bias;
  const float* resid; int ldr; long long sRb, sRh;
  const float* addm;  int ldm; long long sMb, sMh;
  float scale; int act; int K; int zdiv;
};

#define KC 128

__global__ __launch_bounds__(256) void k_gemm(GemmP p) {
  __shared__ __align__(64) uint16_t bsm[64 * KC];
  const int z  = blockIdx.z;
  const int zb = z / p.zdiv, zh = z % p.zdiv;
  const uint16_t* A  = p.A  + zb * p.sAb + zh * p.sAh;
  const uint16_t* Bm = p.Bm + zb * p.sBb + zh * p.sBh;
  const int wid  = threadIdx.x >> 5;
  const int lane = threadIdx.x & 31;
  const int r  = lane & 15;
  const int hh = lane >> 4;
  const int row0 = blockIdx.x * 128 + wid * 16;
  const int col0 = blockIdx.y * 64;
  v8f acc[4] = {};

  for (int k0 = 0; k0 < p.K; k0 += KC) {
    const int kc = (KC < p.K - k0) ? KC : (p.K - k0);
    // ---- stage B strip [64 x kc] (bf16) into LDS ----
#if defined(__HIP_DEVICE_COMPILE__) && __has_builtin(__builtin_amdgcn_tensor_load_to_lds)
    if (wid == 0) {
      const uint16_t* gs = Bm + (size_t)col0 * p.ldb + k0;
      unsigned long long ga = (unsigned long long)(uintptr_t)gs;
      unsigned int la = (unsigned int)(uintptr_t)(&bsm[0]);
      u32x4 g0;
      g0[0] = 1u;                       // count=1 (valid user descriptor)
      g0[1] = la;                       // lds_addr
      g0[2] = (unsigned int)ga;         // global_addr[31:0]
      g0[3] = (unsigned int)((ga >> 32) & 0x1FFFFFFull) | (2u << 30); // addr[56:32] | type=2
      i32x8 g1;
      g1[0] = (1 << 16);                                   // data_size = 2 bytes
      g1[1] = (int)(((unsigned)kc & 0xFFFFu) << 16);       // tensor_dim0[15:0]
      g1[2] = (int)((((unsigned)kc >> 16) & 0xFFFFu) | (64u << 16)); // td0 hi | td1 lo=64
      g1[3] = (int)(((unsigned)kc & 0xFFFFu) << 16);       // td1 hi=0 | tile_dim0=kc
      g1[4] = 64;                                          // tile_dim1=64, tile_dim2=0
      g1[5] = p.ldb;                                       // tensor_dim0_stride[31:0]
      g1[6] = 0; g1[7] = 0;
      i32x4 g2 = {0, 0, 0, 0};
      i32x4 g3 = {0, 0, 0, 0};
#if defined(__clang_major__) && (__clang_major__ >= 23)
      i32x8 g4 = {0, 0, 0, 0, 0, 0, 0, 0};
      __builtin_amdgcn_tensor_load_to_lds(g0, g1, g2, g3, g4, 0);
#else
      __builtin_amdgcn_tensor_load_to_lds(g0, g1, g2, g3, 0);
#endif
#if __has_builtin(__builtin_amdgcn_s_wait_tensorcnt)
      __builtin_amdgcn_s_wait_tensorcnt(0);
#endif
    }
#else
    {
      const uint16_t* gs = Bm + (size_t)col0 * p.ldb + k0;
      const int vec = kc >> 3;
      for (int idx = threadIdx.x; idx < 64 * vec; idx += 256) {
        int rr = idx / vec, cc = idx - rr * vec;
        *(uint4*)(bsm + rr * kc + cc * 8) =
            *(const uint4*)(gs + (size_t)rr * p.ldb + cc * 8);
      }
    }
#endif
    __syncthreads();

    const uint16_t* arow = A + (size_t)(row0 + r) * p.lda + k0 + hh * 8;
    if (k0 + KC < p.K) __builtin_prefetch(arow + KC, 0, 0); // global_prefetch_b8

    for (int kk = 0; kk < kc; kk += 32) {
      v8bf alo = *(const v8bf*)(arow + kk);
      v8bf ahi = *(const v8bf*)(arow + kk + 16);
      v16bf af = __builtin_shufflevector(alo, ahi,
                   0,1,2,3,4,5,6,7,8,9,10,11,12,13,14,15);
      const uint16_t* bbase = bsm + kk + hh * 8;
#pragma unroll
      for (int t = 0; t < 4; ++t) {
        v8bf blo = *(const v8bf*)(bbase + (size_t)(t * 16 + r) * kc);
        v8bf bhi = *(const v8bf*)(bbase + (size_t)(t * 16 + r) * kc + 16);
        v16bf bf = __builtin_shufflevector(blo, bhi,
                     0,1,2,3,4,5,6,7,8,9,10,11,12,13,14,15);
        acc[t] = __builtin_amdgcn_wmma_f32_16x16x32_bf16(
            false, af, false, bf, (short)0, acc[t], false, false);
      }
    }
    __syncthreads();
  }

  // ---- epilogue ----
  const long long co = zb * p.sCb + zh * p.sCh;
#pragma unroll
  for (int t = 0; t < 4; ++t) {
    const int col = col0 + t * 16 + r;
    const float bv = p.bias ? p.bias[col] : 0.0f;
#pragma unroll
    for (int i = 0; i < 8; ++i) {
      const int row = row0 + hh * 8 + i;
      float v = acc[t][i] * p.scale + bv;
      if (p.addm)  v += p.addm[zb * p.sMb + zh * p.sMh + (long long)row * p.ldm + col];
      if (p.act)   v = gelu_ex(v);
      if (p.resid) v += p.resid[zb * p.sRb + zh * p.sRh + (long long)row * p.ldr + col];
      const long long o = co + (long long)row * p.ldc + col;
      if (p.outF) p.outF[o] = v;
      if (p.outB) p.outB[o] = f2b(v);
    }
  }
}

// ---------------- LayerNorm (f32 in -> bf16 out), one wave per row ----------------
template <int C>
__global__ __launch_bounds__(256) void k_ln(const float* __restrict__ in,
                                            const float* __restrict__ g,
                                            const float* __restrict__ b,
                                            uint16_t* __restrict__ ob, int rows) {
  const int wid = threadIdx.x >> 5, lane = threadIdx.x & 31;
  const int row = blockIdx.x * 8 + wid;
  if (row >= rows) return;
  const float* x = in + (size_t)row * C;
  constexpr int NP = C / 32;
  float v[NP];
  float s = 0.f;
#pragma unroll
  for (int i = 0; i < NP; ++i) { v[i] = x[lane + 32 * i]; s += v[i]; }
#pragma unroll
  for (int m = 16; m; m >>= 1) s += __shfl_xor(s, m, 32);
  const float mean = s * (1.0f / C);
  float q = 0.f;
#pragma unroll
  for (int i = 0; i < NP; ++i) { float d = v[i] - mean; q += d * d; }
#pragma unroll
  for (int m = 16; m; m >>= 1) q += __shfl_xor(q, m, 32);
  const float inv = rsqrtf(q * (1.0f / C) + 1e-5f);
  uint16_t* o = ob + (size_t)row * C;
#pragma unroll
  for (int i = 0; i < NP; ++i) {
    const int c = lane + 32 * i;
    o[c] = f2b((v[i] - mean) * inv * g[c] + b[c]);
  }
}

// op row = srow[b,i,:] + scol[b,j,:] computed on the fly, then LN -> bf16
__global__ __launch_bounds__(256) void k_ln_outer(const float* __restrict__ rowt,
                                                  const float* __restrict__ colt,
                                                  const float* __restrict__ g,
                                                  const float* __restrict__ b,
                                                  uint16_t* __restrict__ ob) {
  const int wid = threadIdx.x >> 5, lane = threadIdx.x & 31;
  const long long row = (long long)blockIdx.x * 8 + wid;
  if (row >= (long long)BLLc) return;
  const int bb  = (int)(row / ((long long)Lc * Lc));
  const int rem = (int)(row - (long long)bb * Lc * Lc);
  const int i = rem / Lc, j = rem - i * Lc;
  const float* xr = rowt + (size_t)(bb * Lc + i) * Pc;
  const float* xc = colt + (size_t)(bb * Lc + j) * Pc;
  float v0 = xr[lane] + xc[lane];
  float v1 = xr[lane + 32] + xc[lane + 32];
  float s = v0 + v1;
#pragma unroll
  for (int m = 16; m; m >>= 1) s += __shfl_xor(s, m, 32);
  const float mean = s * (1.0f / Pc);
  float d0 = v0 - mean, d1 = v1 - mean;
  float q = d0 * d0 + d1 * d1;
#pragma unroll
  for (int m = 16; m; m >>= 1) q += __shfl_xor(q, m, 32);
  const float inv = rsqrtf(q * (1.0f / Pc) + 1e-5f);
  uint16_t* o = ob + row * Pc;
  o[lane]      = f2b(d0 * inv * g[lane] + b[lane]);
  o[lane + 32] = f2b(d1 * inv * g[lane + 32] + b[lane + 32]);
}

// bias[b,h,i,j] = sum_p pair_ln[b,i,j,p] * Wb[p,h]  (memory-bound, N=4 -> VALU)
__global__ __launch_bounds__(256) void k_bias_proj(const uint16_t* __restrict__ pln,
                                                   const float* __restrict__ Wb,
                                                   float* __restrict__ biasout) {
  const long long r = (long long)blockIdx.x * 256 + threadIdx.x;
  if (r >= (long long)BLLc) return;
  const int bb  = (int)(r / ((long long)Lc * Lc));
  const int rem = (int)(r - (long long)bb * Lc * Lc);
  const int i = rem / Lc, j = rem - i * Lc;
  const uint16_t* x = pln + r * Pc;
  float s0 = 0, s1 = 0, s2 = 0, s3 = 0;
#pragma unroll 4
  for (int p = 0; p < Pc; ++p) {
    const float xv = b2f(x[p]);
    const float* w = Wb + p * NHc;
    s0 += xv * w[0]; s1 += xv * w[1]; s2 += xv * w[2]; s3 += xv * w[3];
  }
  const long long base = (long long)bb * NHc * Lc * Lc + (long long)i * Lc + j;
  const long long LL = (long long)Lc * Lc;
  biasout[base + 0 * LL] = s0; biasout[base + 1 * LL] = s1;
  biasout[base + 2 * LL] = s2; biasout[base + 3 * LL] = s3;
}

// softmax over 384 cols, one wave per row, bf16 out
__global__ __launch_bounds__(256) void k_softmax(const float* __restrict__ s,
                                                 uint16_t* __restrict__ o, int rows) {
  const int wid = threadIdx.x >> 5, lane = threadIdx.x & 31;
  const int row = blockIdx.x * 8 + wid;
  if (row >= rows) return;
  const float* x = s + (size_t)row * Lc;
  float v[12];
  float mx = -1e30f;
#pragma unroll
  for (int i = 0; i < 12; ++i) { v[i] = x[lane + 32 * i]; mx = fmaxf(mx, v[i]); }
#pragma unroll
  for (int m = 16; m; m >>= 1) mx = fmaxf(mx, __shfl_xor(mx, m, 32));
  float sum = 0.f;
#pragma unroll
  for (int i = 0; i < 12; ++i) { v[i] = __expf(v[i] - mx); sum += v[i]; }
#pragma unroll
  for (int m = 16; m; m >>= 1) sum += __shfl_xor(sum, m, 32);
  const float inv = 1.0f / sum;
  uint16_t* op = o + (size_t)row * Lc;
#pragma unroll
  for (int i = 0; i < 12; ++i) op[lane + 32 * i] = f2b(v[i] * inv);
}

// v [BL,H] bf16 -> vT [B,H,L] bf16 (LDS tiled 32x32)
__global__ __launch_bounds__(256) void k_trans16(const uint16_t* __restrict__ src,
                                                 uint16_t* __restrict__ dst) {
  __shared__ uint16_t tile[32][33];
  const int bz = blockIdx.z;
  const int i0 = blockIdx.x * 32, d0 = blockIdx.y * 32;
  const int tx = threadIdx.x & 31, ty = threadIdx.x >> 5;
#pragma unroll
  for (int k = 0; k < 4; ++k) {
    const int i = ty + k * 8;
    tile[i][tx] = src[(size_t)(bz * Lc + i0 + i) * Hc + d0 + tx];
  }
  __syncthreads();
#pragma unroll
  for (int k = 0; k < 4; ++k) {
    const int d = ty + k * 8;
    dst[(size_t)(bz * Hc + d0 + d) * Lc + i0 + tx] = tile[tx][d];
  }
}

// pair init: rowt[i] + colt[j] + rel[clip(j-i)]
__global__ __launch_bounds__(256) void k_pair_init(const float* __restrict__ rowt,
                                                   const float* __restrict__ colt,
                                                   const float* __restrict__ rel,
                                                   float* __restrict__ pair) {
  const long long t = (long long)blockIdx.x * 256 + threadIdx.x; // BLL*16
  const long long r = t >> 4;
  const int c4 = (int)(t & 15);
  const int bb  = (int)(r / ((long long)Lc * Lc));
  const int rem = (int)(r - (long long)bb * Lc * Lc);
  const int i = rem / Lc, j = rem - i * Lc;
  int d = j - i; d = (d < -MAXRELc) ? -MAXRELc : (d > MAXRELc ? MAXRELc : d); d += MAXRELc;
  const float4 va = ((const float4*)(rowt + (size_t)(bb * Lc + i) * Pc))[c4];
  const float4 vb = ((const float4*)(colt + (size_t)(bb * Lc + j) * Pc))[c4];
  const float4 vr = ((const float4*)(rel + (size_t)d * Pc))[c4];
  float4 o;
  o.x = va.x + vb.x + vr.x; o.y = va.y + vb.y + vr.y;
  o.z = va.z + vb.z + vr.z; o.w = va.w + vb.w + vr.w;
  ((float4*)(pair + r * Pc))[c4] = o;
}

// symb = bf16(0.5*(lg[b,i,j,:] + lg[b,j,i,:]))
__global__ __launch_bounds__(256) void k_symm_b(const float* __restrict__ lg,
                                                uint16_t* __restrict__ symb) {
  const long long t = (long long)blockIdx.x * 256 + threadIdx.x; // BLL*64
  const long long r = t >> 6; const int c = (int)(t & 63);
  const int bb  = (int)(r / ((long long)Lc * Lc));
  const int rem = (int)(r - (long long)bb * Lc * Lc);
  const int i = rem / Lc, j = rem - i * Lc;
  const long long t2 = (((long long)bb * Lc * Lc + (long long)j * Lc + i) << 6) + c;
  symb[t] = f2b(0.5f * (lg[t] + lg[t2]));
}

// in-place symmetrize of final logits (each thread owns (i<=j) pair)
__global__ __launch_bounds__(256) void k_symm_inplace(float* __restrict__ lg) {
  const long long t = (long long)blockIdx.x * 256 + threadIdx.x; // BLL*64
  const long long r = t >> 6; const int c = (int)(t & 63);
  const int bb  = (int)(r / ((long long)Lc * Lc));
  const int rem = (int)(r - (long long)bb * Lc * Lc);
  const int i = rem / Lc, j = rem - i * Lc;
  if (i > j) return;
  const long long t2 = (((long long)bb * Lc * Lc + (long long)j * Lc + i) << 6) + c;
  const float v = 0.5f * (lg[t] + lg[t2]);
  lg[t] = v; lg[t2] = v;
}

// transpose-convert weight: src f32 [K,N] -> dst bf16 [N,Kpad] (zero K padding)
__global__ __launch_bounds__(256) void k_wtrans(const float* __restrict__ src,
                                                uint16_t* __restrict__ dst,
                                                int K, int N, int Kpad) {
  const long long t = (long long)blockIdx.x * 256 + threadIdx.x;
  if (t >= (long long)N * Kpad) return;
  const int n = (int)(t / Kpad), k = (int)(t - (long long)n * Kpad);
  dst[t] = (k < K) ? f2b(src[(size_t)k * N + n]) : (uint16_t)0;
}

// x f32 [BL,48] -> bf16 [BL,64] padded
__global__ __launch_bounds__(256) void k_cvtpad(const float* __restrict__ x,
                                                uint16_t* __restrict__ xb) {
  const long long t = (long long)blockIdx.x * 256 + threadIdx.x; // BL*64
  if (t >= (long long)BLc * 64) return;
  const int r = (int)(t >> 6), c = (int)(t & 63);
  xb[t] = (c < AAc) ? f2b(x[(size_t)r * AAc + c]) : (uint16_t)0;
}

// ---------------- host orchestration ----------------
extern "C" void kernel_launch(void* const* d_in, const int* in_sizes, int n_in,
                              void* d_out, int out_size, void* d_ws, size_t ws_size,
                              hipStream_t stream) {
  if (n_in < 113) return;
  if (ws_size < (size_t)190 * 1024 * 1024) return;

  // ---- inputs (setup_inputs insertion order) ----
  const float* x    = (const float*)d_in[0];
  const float* Wres = (const float*)d_in[1];
  const float* bres = (const float*)d_in[2];
  const float* pos  = (const float*)d_in[3];
  const float* Wpi  = (const float*)d_in[4];
  const float* bpi  = (const float*)d_in[5];
  const float* Wpj  = (const float*)d_in[6];
  const float* bpj  = (const float*)d_in[7];
  const float* rel  = (const float*)d_in[8];
  struct Lw {
    const float *ns_g, *ns_b, *np_g, *np_b, *Wb, *Wq, *Wk, *Wv, *Wo, *bo,
                *ff_g, *ff_b, *W1, *b1, *W2, *b2, *pu_g, *pu_b,
                *pW1, *pb1, *pW2, *pb2, *Wout, *bout;
  };
  Lw lw[NLAY];
  for (int l = 0; l < NLAY; ++l) {
    const float** f = (const float**)&lw[l];
    for (int k = 0; k < 24; ++k) f[k] = (const float*)d_in[9 + l * 24 + k];
  }
  const float* dg_g = (const float*)d_in[105];
  const float* dg_b = (const float*)d_in[106];
  const float* dW1  = (const float*)d_in[107];
  const float* db1  = (const float*)d_in[108];
  const float* dW2  = (const float*)d_in[109];
  const float* db2  = (const float*)d_in[110];
  const float* Wrec = (const float*)d_in[111];
  const float* brec = (const float*)d_in[112];

  // ---- bump allocator on d_ws ----
  char* wp = (char*)d_ws;
  auto alloc = [&](size_t bytes) -> void* {
    char* r = wp; wp += (bytes + 255) & ~(size_t)255; return (void*)r;
  };

  // ---- weight transpose+convert (f32 [K,N] -> bf16 [N,Kpad]) ----
  auto wt = [&](const float* src, int K, int N) -> uint16_t* {
    const int Kpad = (K + 31) & ~31;
    uint16_t* dst = (uint16_t*)alloc((size_t)N * Kpad * 2);
    const long long tot = (long long)N * Kpad;
    k_wtrans<<<(unsigned)((tot + 255) / 256), 256, 0, stream>>>(src, dst, K, N, Kpad);
    return dst;
  };

  uint16_t* WresT = wt(Wres, AAc, Hc);   // [256,64]
  uint16_t* WpiT  = wt(Wpi, Hc, Pc);     // [64,256]
  uint16_t* WpjT  = wt(Wpj, Hc, Pc);
  uint16_t *WqT[NLAY], *WkT[NLAY], *WvT[NLAY], *WoT[NLAY], *W1T[NLAY], *W2T[NLAY],
           *WiT[NLAY], *WjT[NLAY], *pW1T[NLAY], *pW2T[NLAY];
  for (int l = 0; l < NLAY; ++l) {
    WqT[l]  = wt(lw[l].Wq, Hc, Hc);
    WkT[l]  = wt(lw[l].Wk, Hc, Hc);
    WvT[l]  = wt(lw[l].Wv, Hc, Hc);
    WoT[l]  = wt(lw[l].Wo, Hc, Hc);
    W1T[l]  = wt(lw[l].W1, Hc, 4 * Hc);          // [1024,256]
    W2T[l]  = wt(lw[l].W2, 4 * Hc, Hc);          // [256,1024]
    WiT[l]  = wt(lw[l].Wout, Hc, Pc);            // rows 0..255 of Wout
    WjT[l]  = wt(lw[l].Wout + (size_t)Hc * Pc, Hc, Pc); // rows 256..511
    pW1T[l] = wt(lw[l].pW1, Pc, Pc);
    pW2T[l] = wt(lw[l].pW2, Pc, Pc);
  }
  uint16_t* dW1T  = wt(dW1, Pc, Pc);
  uint16_t* dW2T  = wt(dW2, Pc, NBc);
  uint16_t* WrecT = wt(Wrec, NBc, Pc);

  // ---- activations ----
  uint16_t* xb     = (uint16_t*)alloc((size_t)BLc * 64 * 2);
  float*    hinit  = (float*)   alloc((size_t)BLc * Hc * 4);
  uint16_t* hinitb = (uint16_t*)alloc((size_t)BLc * Hc * 2);
  float*    seq    = (float*)   alloc((size_t)BLc * Hc * 4);
  uint16_t* sb     = (uint16_t*)alloc((size_t)BLc * Hc * 2);
  uint16_t* snb    = (uint16_t*)alloc((size_t)BLc * Hc * 2);
  uint16_t* qb     = (uint16_t*)alloc((size_t)BLc * Hc * 2);
  uint16_t* kb     = (uint16_t*)alloc((size_t)BLc * Hc * 2);
  uint16_t* vb     = (uint16_t*)alloc((size_t)BLc * Hc * 2);
  uint16_t* vT     = (uint16_t*)alloc((size_t)BLc * Hc * 2);
  uint16_t* ctxb   = (uint16_t*)alloc((size_t)BLc * Hc * 2);
  uint16_t* ffh    = (uint16_t*)alloc((size_t)BLc * 4 * Hc * 2);
  float*    scoresf= (float*)   alloc((size_t)Bc * NHc * Lc * Lc * 4);
  uint16_t* attnb  = (uint16_t*)alloc((size_t)Bc * NHc * Lc * Lc * 2);
  float*    biasm  = (float*)   alloc((size_t)Bc * NHc * Lc * Lc * 4);
  float*    rowt   = (float*)   alloc((size_t)BLc * Pc * 4);
  float*    colt   = (float*)   alloc((size_t)BLc * Pc * 4);
  float*    pair   = (float*)   alloc((size_t)BLLc * Pc * 4);
  uint16_t* plnb   = (uint16_t*)alloc((size_t)BLLc * Pc * 2);
  uint16_t* t1b    = (uint16_t*)alloc((size_t)BLLc * Pc * 2);

  auto run_gemm = [&](const uint16_t* A, int lda, long long sAb, long long sAh,
                      const uint16_t* Bw, int ldb, long long sBb, long long sBh,
                      float* outF, uint16_t* outB, int ldc, long long sCb, long long sCh,
                      const float* bias, const float* resid, int ldr, long long sRb, long long sRh,
                      const float* addm, int ldm, long long sMb, long long sMh,
                      float scale, int act, int M, int N, int K, int Z, int zdiv) {
    GemmP p;
    p.A = A; p.lda = lda; p.sAb = sAb; p.sAh = sAh;
    p.Bm = Bw; p.ldb = ldb; p.sBb = sBb; p.sBh = sBh;
    p.outF = outF; p.outB = outB; p.ldc = ldc; p.sCb = sCb; p.sCh = sCh;
    p.bias = bias; p.resid = resid; p.ldr = ldr; p.sRb = sRb; p.sRh = sRh;
    p.addm = addm; p.ldm = ldm; p.sMb = sMb; p.sMh = sMh;
    p.scale = scale; p.act = act; p.K = K; p.zdiv = zdiv;
    dim3 g((unsigned)(M / 128), (unsigned)(N / 64), (unsigned)Z);
    k_gemm<<<g, 256, 0, stream>>>(p);
  };
  auto wgemm = [&](const uint16_t* A, int lda, const uint16_t* Bw, int ldb,
                   float* outF, uint16_t* outB, int ldc,
                   const float* bias, const float* resid, int act,
                   int M, int N, int K) {
    run_gemm(A, lda, 0, 0, Bw, ldb, 0, 0, outF, outB, ldc, 0, 0, bias,
             resid, ldc, 0, 0, nullptr, 0, 0, 0, 1.0f, act, M, N, K, 1, 1);
  };

  const long long LL = (long long)Lc * Lc;

  // ---- input embedding ----
  k_cvtpad<<<(BLc * 64) / 256, 256, 0, stream>>>(x, xb);
  // h_init = xb @ WresT + bres + pos   (z over batch; addm = pos with local row idx)
  run_gemm(xb, 64, (long long)Lc * 64, 0,
           WresT, 64, 0, 0,
           hinit, hinitb, Hc, (long long)Lc * Hc, 0,
           bres, nullptr, 0, 0, 0,
           pos, Hc, 0, 0,
           1.0f, 0, Lc, Hc, 64, Bc, 1);
  wgemm(hinitb, Hc, WpiT, Hc, rowt, nullptr, Pc, bpi, nullptr, 0, BLc, Pc, Hc);
  wgemm(hinitb, Hc, WpjT, Hc, colt, nullptr, Pc, bpj, nullptr, 0, BLc, Pc, Hc);
  k_pair_init<<<(unsigned)(((long long)BLLc * 16) / 256), 256, 0, stream>>>(rowt, colt, rel, pair);

  float* lg = (float*)d_out; // use d_out as logits scratch each cycle

  for (int cyc = 0; cyc < NRECc; ++cyc) {
    hipMemcpyAsync(seq, hinit, (size_t)BLc * Hc * sizeof(float),
                   hipMemcpyDeviceToDevice, stream);
    for (int l = 0; l < NLAY; ++l) {
      const Lw& w = lw[l];
      k_ln<256><<<BLc / 8, 256, 0, stream>>>(seq, w.ns_g, w.ns_b, snb, BLc);
      k_ln<64><<<BLLc / 8, 256, 0, stream>>>(pair, w.np_g, w.np_b, plnb, BLLc);
      k_bias_proj<<<(BLLc + 255) / 256, 256, 0, stream>>>(plnb, w.Wb, biasm);
      wgemm(snb, Hc, WqT[l], Hc, nullptr, qb, Hc, nullptr, nullptr, 0, BLc, Hc, Hc);
      wgemm(snb, Hc, WkT[l], Hc, nullptr, kb, Hc, nullptr, nullptr, 0, BLc, Hc, Hc);
      wgemm(snb, Hc, WvT[l], Hc, nullptr, vb, Hc, nullptr, nullptr, 0, BLc, Hc, Hc);
      k_trans16<<<dim3(Lc / 32, Hc / 32, Bc), 256, 0, stream>>>(vb, vT);
      // scores[b,h] = (q @ k^T) * 1/8 + bias[b,h]
      run_gemm(qb, Hc, (long long)Lc * Hc, 64,
               kb, Hc, (long long)Lc * Hc, 64,
               scoresf, nullptr, Lc, (long long)NHc * LL, LL,
               nullptr, nullptr, 0, 0, 0,
               biasm, Lc, (long long)NHc * LL, LL,
               0.125f, 0, Lc, Lc, HDc, Bc * NHc, NHc);
      k_softmax<<<(Bc * NHc * Lc) / 8, 256, 0, stream>>>(scoresf, attnb, Bc * NHc * Lc);
      // ctx[b,:,h*64..] = attn[b,h] @ v[b,:,h*64..]
      run_gemm(attnb, Lc, (long long)NHc * LL, LL,
               vT, Lc, (long long)Hc * Lc, (long long)64 * Lc,
               nullptr, ctxb, Hc, (long long)Lc * Hc, 64,
               nullptr, nullptr, 0, 0, 0, nullptr, 0, 0, 0,
               1.0f, 0, Lc, HDc, Lc, Bc * NHc, NHc);
      // s = ctx @ Wo + bo + seq
      wgemm(ctxb, Hc, WoT[l], Hc, seq, nullptr, Hc, w.bo, seq, 0, BLc, Hc, Hc);
      // FFN
      k_ln<256><<<BLc / 8, 256, 0, stream>>>(seq, w.ff_g, w.ff_b, snb, BLc);
      wgemm(snb, Hc, W1T[l], Hc, nullptr, ffh, 4 * Hc, w.b1, nullptr, 1, BLc, 4 * Hc, Hc);
      wgemm(ffh, 4 * Hc, W2T[l], 4 * Hc, seq, sb, Hc, w.b2, seq, 0, BLc, Hc, 4 * Hc);
      // outer product terms + pair transition
      wgemm(sb, Hc, WiT[l], Hc, rowt, nullptr, Pc, w.bout, nullptr, 0, BLc, Pc, Hc);
      wgemm(sb, Hc, WjT[l], Hc, colt, nullptr, Pc, nullptr, nullptr, 0, BLc, Pc, Hc);
      k_ln_outer<<<BLLc / 8, 256, 0, stream>>>(rowt, colt, w.pu_g, w.pu_b, plnb);
      wgemm(plnb, Pc, pW1T[l], Pc, nullptr, t1b, Pc, w.pb1, nullptr, 1, BLLc, Pc, Pc);
      wgemm(t1b, Pc, pW2T[l], Pc, pair, nullptr, Pc, w.pb2, pair, 0, BLLc, Pc, Pc);
    }
    // distogram
    k_ln<64><<<BLLc / 8, 256, 0, stream>>>(pair, dg_g, dg_b, plnb, BLLc);
    wgemm(plnb, Pc, dW1T, Pc, nullptr, t1b, Pc, db1, nullptr, 1, BLLc, Pc, Pc);
    wgemm(t1b, Pc, dW2T, Pc, lg, nullptr, NBc, db2, nullptr, 0, BLLc, NBc, Pc);
    if (cyc < NRECc - 1) {
      k_symm_b<<<(unsigned)(((long long)BLLc * 64) / 256), 256, 0, stream>>>(lg, plnb);
      wgemm(plnb, NBc, WrecT, NBc, pair, nullptr, Pc, brec, pair, 0, BLLc, Pc, NBc);
    } else {
      k_symm_inplace<<<(unsigned)(((long long)BLLc * 64) / 256), 256, 0, stream>>>(lg);
    }
  }
  (void)in_sizes; (void)out_size;
}
